// RobustPPRGo_15290083574244
// MI455X (gfx1250) — matrix operand order repaired and checked by reading, not compiled, not run on hardware
//
#include <hip/hip_runtime.h>
#include <hip/hip_bf16.h>
#include <math.h>

// ---------- problem constants (from reference) ----------
#define NN   200000   // nodes
#define FF   512      // input features
#define HH   512      // hidden
#define CC   64       // classes
#define BB   8192     // batch rows
#define NNZv 64       // neighbors per row
#define KSEL 32       // soft medoid k

typedef __attribute__((ext_vector_type(16))) __bf16        v16bf;
typedef __attribute__((ext_vector_type(8)))  float          v8f;
typedef __attribute__((ext_vector_type(8)))  unsigned short u16x8;
typedef __attribute__((ext_vector_type(4)))  int            v4i;

// ---------- CDNA5 async global->LDS path (guarded; fallback = sync copy) ----
#if defined(__has_builtin)
#if __has_builtin(__builtin_amdgcn_global_load_async_to_lds_b128)
#define HAVE_ASYNC_LDS 1
#endif
#endif
#ifndef HAVE_ASYNC_LDS
#define HAVE_ASYNC_LDS 0
#endif

__device__ __forceinline__ void cp16(const unsigned short* g, unsigned short* l) {
#if HAVE_ASYNC_LDS
    __builtin_amdgcn_global_load_async_to_lds_b128(
        (__attribute__((address_space(1))) v4i*)g,
        (__attribute__((address_space(3))) v4i*)l, 0, 0);
#else
    *(u16x8*)l = *(const u16x8*)g;
#endif
}

__device__ __forceinline__ void wait_async0() {
#if HAVE_ASYNC_LDS
#if __has_builtin(__builtin_amdgcn_s_wait_asynccnt)
    __builtin_amdgcn_s_wait_asynccnt(0);
#else
    asm volatile("s_wait_asynccnt 0x0" ::: "memory");
#endif
#endif
}

__device__ __forceinline__ unsigned short f2bf(float f) {
    union { float f; unsigned int u; } x; x.f = f;
    unsigned int u = x.u;
    unsigned int lsb = (u >> 16) & 1u;
    u += 0x7fffu + lsb;                 // round to nearest even
    return (unsigned short)(u >> 16);
}

// ---------- f32 -> bf16 conversion (bandwidth pass) ----------
__global__ void cvt_f32_to_bf16(const float* __restrict__ in,
                                unsigned short* __restrict__ out, long n) {
    long i = (long)blockIdx.x * blockDim.x + threadIdx.x;
    long stride = (long)gridDim.x * blockDim.x;
    for (; i < n; i += stride) out[i] = f2bf(in[i]);
}

// ---------- f32 [Kd][Nd] -> bf16 transposed [Nd][Kd] (weights, one-shot) ----
__global__ void transpose_w_bf16(const float* __restrict__ W,
                                 unsigned short* __restrict__ WT,
                                 int Kd, int Nd) {
    int idx = blockIdx.x * blockDim.x + threadIdx.x;
    if (idx < Kd * Nd) {
        int k = idx / Nd, n = idx - k * Nd;
        WT[(size_t)n * Kd + k] = f2bf(W[idx]);
    }
}

// ---------- bf16 WMMA GEMM: Out[M,Nd] = act(A[M,Kd] @ W[Kd,Nd]) ----------
// W passed pre-transposed as WT[Nd][Kd] so all staging is contiguous 16B chunks.
// Block: 256 threads (8 wave32). Tile: BM=128, BN=64, BK=32, double-buffered
// async global->LDS staging for both A and B tiles.
template<bool RELU, bool OUT_BF16>
__global__ void __launch_bounds__(256)
gemm_bf16_wmma(const unsigned short* __restrict__ A,    // [M][Kd] bf16 row-major
               const unsigned short* __restrict__ WT,   // [Nd][Kd] bf16
               void* __restrict__ Out, int M, int Kd, int Nd) {
    __shared__ __align__(16) unsigned short lA[2][128 * 32];  // [m][k]
    __shared__ __align__(16) unsigned short lBt[2][64 * 32];  // [n][k]

    const int tid  = threadIdx.x;
    const int wave = tid >> 5;
    const int lane = tid & 31;
    const int g    = lane >> 4;   // half-wave group (ISA K-split)
    const int r    = lane & 15;
    const int m0   = blockIdx.x * 128;
    const int n0   = blockIdx.y * 64;

    union Frag { u16x8 h[2]; v16bf v; };
    union Acc  { v8f v; float f[8]; };

    // A tile: 512 x 16B chunks -> 2 per thread
    // chunk c: row = c>>2, k-offset = (c&3)*8
    int aRow0 = (tid * 2) >> 2;          // both chunks of a thread share a row
    int aGRow = m0 + aRow0; if (aGRow >= M) aGRow = M - 1;   // clamp (stores guarded)

    // B tile: 256 x 16B chunks -> 1 per thread
    const int bRow = tid >> 2;           // 0..63 (n within tile)
    const int bKo  = (tid & 3) * 8;

    auto stageA = [&](int kk, int buf) {
        #pragma unroll
        for (int i = 0; i < 2; ++i) {
            int c  = tid * 2 + i;
            int ko = (c & 3) * 8;
            cp16(A + (size_t)aGRow * Kd + kk + ko, &lA[buf][(c >> 2) * 32 + ko]);
        }
    };
    auto stageB = [&](int kk, int buf) {
        cp16(WT + (size_t)(n0 + bRow) * Kd + kk + bKo, &lBt[buf][bRow * 32 + bKo]);
    };

    Acc acc[4] = {};

    stageA(0, 0);
    stageB(0, 0);

    const int nk = Kd / 32;
    for (int ik = 0; ik < nk; ++ik) {
        wait_async0();
        __syncthreads();
        if (ik + 1 < nk) {
            stageA((ik + 1) * 32, (ik + 1) & 1);
            stageB((ik + 1) * 32, (ik + 1) & 1);
        }
        const int buf = ik & 1;

        // A fragment per ISA 16-bit 16x32 layout:
        // lanes 0-15 (g=0): K 0-7 then 16-23 ; lanes 16-31 (g=1): K 8-15 then 24-31
        Frag fa;
        const unsigned short* ap = &lA[buf][(wave * 16 + r) * 32];
        fa.h[0] = *(const u16x8*)(ap + 8 * g);
        fa.h[1] = *(const u16x8*)(ap + 16 + 8 * g);

        #pragma unroll
        for (int t = 0; t < 4; ++t) {
            // B fragment: col fixed = 16t+r, K = 8g..8g+7 then 16+8g..16+8g+7 (matches A)
            Frag fb;
            const unsigned short* bp = &lBt[buf][(t * 16 + r) * 32 + 8 * g];
            fb.h[0] = *(const u16x8*)(bp);
            fb.h[1] = *(const u16x8*)(bp + 16);
            acc[t].v = __builtin_amdgcn_wmma_f32_16x16x32_bf16(
                false, fa.v, false, fb.v, (short)0, acc[t].v, false, false);
        }
        // no trailing barrier: this buffer is only overwritten after the next
        // iteration's wait+barrier, which orders all waves' reads before it.
    }

    // C/D layout: VGPR rr, lane -> M = rr + 8g, N = lane%16
    #pragma unroll
    for (int t = 0; t < 4; ++t) {
        #pragma unroll
        for (int rr = 0; rr < 8; ++rr) {
            int gm = m0 + wave * 16 + rr + 8 * g;
            if (gm < M) {
                int gn = n0 + t * 16 + r;
                float v = acc[t].f[rr];
                if (RELU) v = v > 0.f ? v : 0.f;
                if (OUT_BF16)
                    ((unsigned short*)Out)[(size_t)gm * Nd + gn] = f2bf(v);
                else
                    ((float*)Out)[(size_t)gm * Nd + gn] = v;
            }
        }
    }
}

// ---------- per-row soft-k-medoid (one block per PPR row) ----------
__global__ void __launch_bounds__(256)
medoid_kernel(const float* __restrict__ logits, const float* __restrict__ pw,
              const int* __restrict__ pidx, float* __restrict__ out) {
    __shared__ __align__(16) float nb[NNZv][CC];   // gathered neighbor logits
    __shared__ float w[NNZv], sqn[NNZv], dist[KSEL], sv[KSEL];
    __shared__ int   cpos[KSEL];
    __shared__ float rowsum_s;

    const int b = blockIdx.x;
    const int t = threadIdx.x;

    if (t < NNZv) w[t] = pw[(size_t)b * NNZv + t];
    if (t < KSEL) dist[t] = 0.f;

    // gather neigh = logits[ppr_idx[b]]: thread -> (n = t>>2, 16 cols)
    {
        int n = t >> 2, cb = (t & 3) * 16;
        int gi = pidx[(size_t)b * NNZv + n];
        const float4* src = (const float4*)(logits + (size_t)gi * CC + cb);
        float4* dst = (float4*)(&nb[n][cb]);
        #pragma unroll
        for (int j = 0; j < 4; ++j) dst[j] = src[j];
    }
    __syncthreads();

    // squared norms + rank-based top-K selection (ties broken by index -> ranks unique)
    if (t < NNZv) {
        float s = 0.f;
        #pragma unroll 8
        for (int c = 0; c < CC; ++c) s += nb[t][c] * nb[t][c];
        sqn[t] = s;
        float wi = w[t];
        int rank = 0;
        for (int j = 0; j < NNZv; ++j) {
            float wj = w[j];
            rank += (wj > wi) || (wj == wi && j < t);
        }
        if (rank < KSEL) cpos[rank] = t;
    }
    if (t == 0) {
        float rs = 0.f;
        for (int j = 0; j < NNZv; ++j) rs += w[j];
        rowsum_s = rs;
    }
    __syncthreads();

    // dist[k] = sum_n w[n] * sqrt(max(|c_k|^2 + |x_n|^2 - 2<c_k,x_n>, 0))
    // 32x64 = 2048 (k,n) pairs, 8 per thread, LDS float atomics (ds_add_f32)
    #pragma unroll
    for (int q = 0; q < 8; ++q) {
        int p  = t + 256 * q;
        int k  = p >> 6, n = p & 63;
        int ck = cpos[k];
        float d = 0.f;
        #pragma unroll 8
        for (int c = 0; c < CC; ++c) d += nb[ck][c] * nb[n][c];
        float l2sq = sqn[ck] + sqn[n] - 2.f * d;
        float l2 = sqrtf(l2sq > 0.f ? l2sq : 0.f);
        atomicAdd(&dist[k], w[n] * l2);
    }
    __syncthreads();

    // softmax over 32 candidates + weight correction (tiny, serial)
    if (t == 0) {
        float rs  = rowsum_s;
        float inv = 1.f / rs;              // TEMP = 1
        float mx  = -3.4e38f;
        float dd[KSEL];
        for (int k = 0; k < KSEL; ++k) {
            float tw = w[cpos[k]];
            float d  = (tw == 0.f) ? 3.4e38f : dist[k];
            float v  = -d * inv;
            dd[k] = v;
            mx = v > mx ? v : mx;
        }
        float se = 0.f;
        for (int k = 0; k < KSEL; ++k) { dd[k] = __expf(dd[k] - mx); se += dd[k]; }
        float s2 = 0.f;
        for (int k = 0; k < KSEL; ++k) { dd[k] = dd[k] / se * w[cpos[k]]; s2 += dd[k]; }
        float scale = rs / s2;             // fold row_sum into weights
        for (int k = 0; k < KSEL; ++k) sv[k] = dd[k] * scale;
    }
    __syncthreads();

    // out[b][c] = sum_k sv[k] * cand[k][c]
    if (t < CC) {
        float o = 0.f;
        #pragma unroll
        for (int k = 0; k < KSEL; ++k) o += sv[k] * nb[cpos[k]][t];
        out[(size_t)b * CC + t] = o;
    }
}

// ---------- launch ----------
extern "C" void kernel_launch(void* const* d_in, const int* in_sizes, int n_in,
                              void* d_out, int out_size, void* d_ws, size_t ws_size,
                              hipStream_t stream) {
    (void)in_sizes; (void)n_in; (void)out_size; (void)ws_size;
    const float* X    = (const float*)d_in[0];
    const float* pw   = (const float*)d_in[1];
    const int*   pidx = (const int*)d_in[2];
    const float* W0   = (const float*)d_in[3];
    const float* W1   = (const float*)d_in[4];
    const float* W2   = (const float*)d_in[5];
    float* out = (float*)d_out;

    char* ws = (char*)d_ws;
    size_t off = 0;
    unsigned short* Xb   = (unsigned short*)(ws + off); off += (size_t)NN * FF * 2;
    unsigned short* W0t  = (unsigned short*)(ws + off); off += (size_t)FF * HH * 2;
    unsigned short* W1t  = (unsigned short*)(ws + off); off += (size_t)HH * HH * 2;
    unsigned short* W2t  = (unsigned short*)(ws + off); off += (size_t)HH * CC * 2;
    unsigned short* H0   = (unsigned short*)(ws + off); off += (size_t)NN * HH * 2;
    unsigned short* H1   = (unsigned short*)(ws + off); off += (size_t)NN * HH * 2;
    float* logitsbuf     = (float*)(ws + off);

    cvt_f32_to_bf16<<<8192, 256, 0, stream>>>(X, Xb, (long)NN * FF);
    transpose_w_bf16<<<(FF * HH + 255) / 256, 256, 0, stream>>>(W0, W0t, FF, HH);
    transpose_w_bf16<<<(HH * HH + 255) / 256, 256, 0, stream>>>(W1, W1t, HH, HH);
    transpose_w_bf16<<<(HH * CC + 255) / 256, 256, 0, stream>>>(W2, W2t, HH, CC);

    dim3 blk(256);
    dim3 g12((NN + 127) / 128, HH / 64);
    gemm_bf16_wmma<true,  true ><<<g12, blk, 0, stream>>>(Xb, W0t, (void*)H0, NN, FF, HH);
    gemm_bf16_wmma<true,  true ><<<g12, blk, 0, stream>>>(H0, W1t, (void*)H1, NN, HH, HH);
    dim3 g3((NN + 127) / 128, CC / 64);
    gemm_bf16_wmma<false, false><<<g3,  blk, 0, stream>>>(H1, W2t, (void*)logitsbuf, NN, HH, CC);

    medoid_kernel<<<BB, 256, 0, stream>>>(logitsbuf, pw, pidx, out);
}
